// IWRMSequential_37538014167835
// MI455X (gfx1250) — matrix operand order repaired and checked by reading, not compiled, not run on hardware
//
#include <hip/hip_runtime.h>

typedef unsigned short u16;
typedef unsigned int   u32;
typedef __attribute__((ext_vector_type(16))) __bf16 v16bf;
typedef __attribute__((ext_vector_type(8)))  float  v8f;

#define NN 512
#define MATS (NN * NN)
#define MATB ((size_t)MATS * sizeof(u16))

union Frag {
  v16bf v;
  uint4 q[2];
};

__device__ __forceinline__ u16 f2bf(float x) {
  u32 u = __builtin_bit_cast(u32, x);
  u32 r = u + 0x7FFFu + ((u >> 16) & 1u);  // round-to-nearest-even
  return (u16)(r >> 16);
}
__device__ __forceinline__ float bf2f(u16 h) {
  return __builtin_bit_cast(float, (u32)h << 16);
}
__device__ __forceinline__ void split2(float x, u16* hi, u16* lo) {
  u16 h = f2bf(x);
  *hi = h;
  *lo = f2bf(x - bf2f(h));
}

// 16x16 fp32 tile of P @ Q^T (both row-major [*,512]) with split-bf16
// (hi+lo) operands: acc = sum_k Ph*Qh + Ph*Ql + Pl*Qh  (~fp32 accuracy).
// A-frag (16x32): lane m=lane&15, c=lane>>4 -> row iBase+m, dwords c*4.. and 8+c*4..
// B-frag (32x16 = Q^T tile): lane n=lane&15 -> row jBase+n, dwords c*4.. and 8+c*4..
// Both are two global_load_b128 per operand per k-step.
__device__ __forceinline__ v8f mm_tile(const u16* __restrict__ Ph, const u16* __restrict__ Pl,
                                       const u16* __restrict__ Qh, const u16* __restrict__ Ql,
                                       int iBase, int jBase, int lane)
{
  const int m = lane & 15;
  const int c = lane >> 4;
  v8f acc = {};
  const u16* pH = Ph + (size_t)(iBase + m) * NN;
  const u16* pL = Pl + (size_t)(iBase + m) * NN;
  const u16* qH = Qh + (size_t)(jBase + m) * NN;
  const u16* qL = Ql + (size_t)(jBase + m) * NN;
  for (int kb = 0; kb < NN; kb += 32) {
    Frag ah, al, bh, bl;
    const uint4* a4h = (const uint4*)(pH + kb);
    const uint4* a4l = (const uint4*)(pL + kb);
    const uint4* b4h = (const uint4*)(qH + kb);
    const uint4* b4l = (const uint4*)(qL + kb);
    ah.q[0] = a4h[c];     ah.q[1] = a4h[2 + c];
    al.q[0] = a4l[c];     al.q[1] = a4l[2 + c];
    bh.q[0] = b4h[c];     bh.q[1] = b4h[2 + c];
    bl.q[0] = b4l[c];     bl.q[1] = b4l[2 + c];
    acc = __builtin_amdgcn_wmma_f32_16x16x32_bf16(false, ah.v, false, bh.v, (short)0, acc, false, false);
    acc = __builtin_amdgcn_wmma_f32_16x16x32_bf16(false, ah.v, false, bl.v, (short)0, acc, false, false);
    acc = __builtin_amdgcn_wmma_f32_16x16x32_bf16(false, al.v, false, bh.v, (short)0, acc, false, false);
  }
  return acc;
}

// ---------------- prep: fp32 -> split bf16 (+ transposes for W1, W2) ----------
__global__ void prep_kernel(const float* __restrict__ x, const float* __restrict__ W1,
                            const float* __restrict__ W2,
                            u16* Xh, u16* Xl, u16* W1th, u16* W1tl,
                            u16* W2th, u16* W2tl, u16* W2rh, u16* W2rl)
{
  int idx = blockIdx.x * blockDim.x + threadIdx.x;
  if (idx >= MATS) return;
  int r = idx >> 9, c = idx & (NN - 1);
  int t = c * NN + r;  // transposed position
  split2(x[idx], &Xh[idx], &Xl[idx]);
  split2(W1[idx], &W1th[t], &W1tl[t]);     // W1t[n,k] = W1[k,n]
  float w2 = W2[idx];
  split2(w2, &W2th[t], &W2tl[t]);          // W2t[n,k] = W2[k,n]
  split2(w2, &W2rh[idx], &W2rl[idx]);      // row-major copy for G2 @ W2^T
}

// ---------------- H1 = relu(X @ W1 + b1) -------------------------------------
__global__ void gemm1_kernel(const u16* __restrict__ Xh, const u16* __restrict__ Xl,
                             const u16* __restrict__ W1th, const u16* __restrict__ W1tl,
                             const float* __restrict__ b1,
                             u16* H1h, u16* H1l)
{
  int wid  = (blockIdx.x * blockDim.x + threadIdx.x) >> 5;
  int lane = threadIdx.x & 31;
  int iBase = (wid >> 5) * 16, jBase = (wid & 31) * 16;
  v8f acc = mm_tile(Xh, Xl, W1th, W1tl, iBase, jBase, lane);
  int m = lane & 15, c = lane >> 4;
  int col = jBase + m;
  float bias = b1[col];
#pragma unroll
  for (int r = 0; r < 8; ++r) {
    int row = iBase + c * 8 + r;
    float h = acc[r] + bias;
    h = h > 0.f ? h : 0.f;
    split2(h, &H1h[row * NN + col], &H1l[row * NN + col]);
  }
}

// ---------------- H2 = relu(H1 @ W2 + b2),  G2 = w3 * 1[z2>0] ----------------
__global__ void gemm2_kernel(const u16* __restrict__ H1h, const u16* __restrict__ H1l,
                             const u16* __restrict__ W2th, const u16* __restrict__ W2tl,
                             const float* __restrict__ b2, const float* __restrict__ w3,
                             u16* H2h, u16* H2l, u16* G2h, u16* G2l)
{
  int wid  = (blockIdx.x * blockDim.x + threadIdx.x) >> 5;
  int lane = threadIdx.x & 31;
  int iBase = (wid >> 5) * 16, jBase = (wid & 31) * 16;
  v8f acc = mm_tile(H1h, H1l, W2th, W2tl, iBase, jBase, lane);
  int m = lane & 15, c = lane >> 4;
  int col = jBase + m;
  float bias = b2[col];
  float w3c  = w3[col];
#pragma unroll
  for (int r = 0; r < 8; ++r) {
    int row = iBase + c * 8 + r;
    float z = acc[r] + bias;
    float h = z > 0.f ? z : 0.f;
    float g = z > 0.f ? w3c : 0.f;
    split2(h, &H2h[row * NN + col], &H2l[row * NN + col]);
    split2(g, &G2h[row * NN + col], &G2l[row * NN + col]);
  }
}

// ---------------- G1 = (G2 @ W2^T) * 1[h1>0] ---------------------------------
__global__ void gemm3_kernel(const u16* __restrict__ G2h, const u16* __restrict__ G2l,
                             const u16* __restrict__ W2rh, const u16* __restrict__ W2rl,
                             const u16* __restrict__ H1h, const u16* __restrict__ H1l,
                             u16* G1h, u16* G1l)
{
  int wid  = (blockIdx.x * blockDim.x + threadIdx.x) >> 5;
  int lane = threadIdx.x & 31;
  int iBase = (wid >> 5) * 16, jBase = (wid & 31) * 16;
  // (G2 @ W2^T)[i,n] = sum_k G2[i,k] * W2[n,k]  -> P=G2, Q=W2 (row-major)
  v8f acc = mm_tile(G2h, G2l, W2rh, W2rl, iBase, jBase, lane);
  int m = lane & 15, c = lane >> 4;
  int col = jBase + m;
#pragma unroll
  for (int r = 0; r < 8; ++r) {
    int row = iBase + c * 8 + r;
    float h1 = bf2f(H1h[row * NN + col]) + bf2f(H1l[row * NN + col]);
    float g  = h1 > 0.f ? acc[r] : 0.f;
    split2(g, &G1h[row * NN + col], &G1l[row * NN + col]);
  }
}

// ---------------- out = H2 @ w3 + b3 (one wave per row) ----------------------
__global__ void outvec_kernel(const u16* __restrict__ H2h, const u16* __restrict__ H2l,
                              const float* __restrict__ w3, const float* __restrict__ b3,
                              float* __restrict__ out)
{
  int wid  = (blockIdx.x * blockDim.x + threadIdx.x) >> 5;
  int lane = threadIdx.x & 31;
  if (wid >= NN) return;
  float acc = 0.f;
  for (int k = lane; k < NN; k += 32) {
    float h = bf2f(H2h[wid * NN + k]) + bf2f(H2l[wid * NN + k]);
    acc += h * w3[k];
  }
#pragma unroll
  for (int off = 16; off > 0; off >>= 1)
    acc += __shfl_xor(acc, off, 32);
  if (lane == 0) out[wid] = acc + b3[0];
}

// ---------------- gram = 1 + Sh2 + (1+Sh1)*Sg2 + (1+Sxx)*Sg1 -----------------
__global__ void gram_kernel(const u16* __restrict__ Xh,  const u16* __restrict__ Xl,
                            const u16* __restrict__ H1h, const u16* __restrict__ H1l,
                            const u16* __restrict__ H2h, const u16* __restrict__ H2l,
                            const u16* __restrict__ G1h, const u16* __restrict__ G1l,
                            const u16* __restrict__ G2h, const u16* __restrict__ G2l,
                            float* __restrict__ gram)
{
  int wid  = (blockIdx.x * blockDim.x + threadIdx.x) >> 5;
  int lane = threadIdx.x & 31;
  int iBase = (wid >> 5) * 16, jBase = (wid & 31) * 16;
  v8f sxx = mm_tile(Xh,  Xl,  Xh,  Xl,  iBase, jBase, lane);
  v8f sh1 = mm_tile(H1h, H1l, H1h, H1l, iBase, jBase, lane);
  v8f sh2 = mm_tile(H2h, H2l, H2h, H2l, iBase, jBase, lane);
  v8f sg1 = mm_tile(G1h, G1l, G1h, G1l, iBase, jBase, lane);
  v8f sg2 = mm_tile(G2h, G2l, G2h, G2l, iBase, jBase, lane);
  int m = lane & 15, c = lane >> 4;
  int col = jBase + m;
#pragma unroll
  for (int r = 0; r < 8; ++r) {
    int row = iBase + c * 8 + r;
    float v = 1.f + sh2[r] + (1.f + sh1[r]) * sg2[r] + (1.f + sxx[r]) * sg1[r];
    gram[row * NN + col] = v;
  }
}

extern "C" void kernel_launch(void* const* d_in, const int* in_sizes, int n_in,
                              void* d_out, int out_size, void* d_ws, size_t ws_size,
                              hipStream_t stream)
{
  (void)in_sizes; (void)n_in; (void)out_size;
  const float* x  = (const float*)d_in[0];
  const float* W1 = (const float*)d_in[1];
  const float* b1 = (const float*)d_in[2];
  const float* W2 = (const float*)d_in[3];
  const float* b2 = (const float*)d_in[4];
  const float* w3 = (const float*)d_in[5];
  const float* b3 = (const float*)d_in[6];
  float* out  = (float*)d_out;
  float* gram = out + NN;

  if (ws_size < 16 * MATB) return;  // need 8 MB of bf16 scratch
  char* ws = (char*)d_ws;
  u16* Xh   = (u16*)(ws +  0 * MATB);
  u16* Xl   = (u16*)(ws +  1 * MATB);
  u16* W1th = (u16*)(ws +  2 * MATB);
  u16* W1tl = (u16*)(ws +  3 * MATB);
  u16* W2th = (u16*)(ws +  4 * MATB);
  u16* W2tl = (u16*)(ws +  5 * MATB);
  u16* W2rh = (u16*)(ws +  6 * MATB);
  u16* W2rl = (u16*)(ws +  7 * MATB);
  u16* H1h  = (u16*)(ws +  8 * MATB);
  u16* H1l  = (u16*)(ws +  9 * MATB);
  u16* H2h  = (u16*)(ws + 10 * MATB);
  u16* H2l  = (u16*)(ws + 11 * MATB);
  u16* G2h  = (u16*)(ws + 12 * MATB);
  u16* G2l  = (u16*)(ws + 13 * MATB);
  u16* G1h  = (u16*)(ws + 14 * MATB);
  u16* G1l  = (u16*)(ws + 15 * MATB);

  prep_kernel<<<MATS / 256, 256, 0, stream>>>(x, W1, W2, Xh, Xl, W1th, W1tl,
                                              W2th, W2tl, W2rh, W2rl);
  gemm1_kernel<<<128, 256, 0, stream>>>(Xh, Xl, W1th, W1tl, b1, H1h, H1l);
  gemm2_kernel<<<128, 256, 0, stream>>>(H1h, H1l, W2th, W2tl, b2, w3,
                                        H2h, H2l, G2h, G2l);
  gemm3_kernel<<<128, 256, 0, stream>>>(G2h, G2l, W2rh, W2rl, H1h, H1l, G1h, G1l);
  outvec_kernel<<<64, 256, 0, stream>>>(H2h, H2l, w3, b3, out);
  gram_kernel<<<128, 256, 0, stream>>>(Xh, Xl, H1h, H1l, H2h, H2l,
                                       G1h, G1l, G2h, G2l, gram);
}
